// SphereConv_3118146257532
// MI455X (gfx1250) — matrix operand order repaired and checked by reading, not compiled, not run on hardware
//
#include <hip/hip_runtime.h>
#include <stdint.h>

// ---------------- problem constants ----------------
#define B_    8
#define V_    49152
#define FIN   64
#define FOUT  64
#define KC    5
#define E_    (V_ * 9)
#define ROWS  (B_ * V_)                 // 393216 rows of the big GEMM
#define NEL   ((size_t)ROWS * FIN)      // 25,165,824 elements per basis matrix
#define KDIM  (KC * FIN)                // 320 reduction dim of fused GEMM
#define WPITCH 328                      // LDS row pitch (bf16) for weights

typedef __attribute__((ext_vector_type(16))) __bf16 v16bf;
typedef __attribute__((ext_vector_type(8)))  float  v8f;

union Frag {
  uint4        u4[2];
  unsigned int u[8];
  v16bf        v;
};

__device__ __forceinline__ unsigned short f2bf(float f) {
  union { float f; unsigned int u; } c; c.f = f;
  unsigned int u = c.u;
  u += 0x7FFFu + ((u >> 16) & 1u);      // round-to-nearest-even
  return (unsigned short)(u >> 16);
}
__device__ __forceinline__ float bf2f(unsigned short h) {
  union { unsigned int u; float f; } c; c.u = ((unsigned int)h) << 16;
  return c.f;
}

// ---------------- 1) transpose [B,Fin,V] -> bf16 [B,V,Fin] (T0) ----------------
__global__ void k_transpose_bf16(const float* __restrict__ x,
                                 unsigned short* __restrict__ t0) {
  __shared__ float tile[64][65];
  int tid = threadIdx.x;
  int tx = tid & 63, ty = tid >> 6;     // 64 x 4
  int v0 = blockIdx.x * 64;
  int b  = blockIdx.y;
  const float* xb = x + (size_t)b * FIN * V_;
  #pragma unroll
  for (int r = 0; r < 16; ++r) {
    int i = ty * 16 + r;
    tile[i][tx] = xb[(size_t)i * V_ + v0 + tx];   // coalesced over v
  }
  __syncthreads();
  unsigned short* tb = t0 + ((size_t)b * V_ + v0) * FIN;
  #pragma unroll
  for (int r = 0; r < 16; ++r) {
    int wv = ty * 16 + r;
    tb[(size_t)wv * FIN + tx] = f2bf(tile[tx][wv]); // coalesced over i
  }
}

// ---------------- 2) SpMM: y[b,row,:] += val * xprev[b,col,:] ----------------
// One wave per edge (lanes = 2 channels each); edges uniform within a wave so
// rows/cols/vals loads are broadcast. Atomic f32 adds land in L2 (y fits).
__global__ void k_spmm(const unsigned short* __restrict__ xprev,
                       const int* __restrict__ rows, const int* __restrict__ cols,
                       const float* __restrict__ vals, float* __restrict__ y) {
  int lane = threadIdx.x & 31;
  int wave = threadIdx.x >> 5;
  long long wi0 = ((long long)blockIdx.x * 8 + wave) * 8;
  for (int j = 0; j < 8; ++j) {
    long long wi = wi0 + j;
    int e = (int)(wi % E_);
    int b = (int)(wi / E_);
    int r = rows[e], c = cols[e];
    float w = vals[e];
    unsigned int p = *(const unsigned int*)(xprev + ((size_t)b * V_ + c) * FIN + lane * 2);
    float v0 = bf2f((unsigned short)(p & 0xFFFFu));
    float v1 = bf2f((unsigned short)(p >> 16));
    float* yr = y + ((size_t)b * V_ + r) * FIN + lane * 2;
    atomicAdd(yr,     w * v0);
    atomicAdd(yr + 1, w * v1);
  }
}

// ---------------- 3) Chebyshev combine: T_k = bf16(2*y - T_{k-2}) ----------------
__global__ void k_cheb(const float* __restrict__ y,
                       const unsigned short* __restrict__ tprev2,
                       unsigned short* __restrict__ tk, int doCombine) {
  size_t i4 = (size_t)blockIdx.x * 256 + threadIdx.x;
  float4 yv = ((const float4*)y)[i4];
  float4 r = yv;
  if (doCombine) {
    ushort4 p = ((const ushort4*)tprev2)[i4];
    r.x = 2.f * yv.x - bf2f(p.x);
    r.y = 2.f * yv.y - bf2f(p.y);
    r.z = 2.f * yv.z - bf2f(p.z);
    r.w = 2.f * yv.w - bf2f(p.w);
  }
  ushort4 o;
  o.x = f2bf(r.x); o.y = f2bf(r.y); o.z = f2bf(r.z); o.w = f2bf(r.w);
  ((ushort4*)tk)[i4] = o;
}

// ---------------- 4) fused GEMM: out[b,o,v] = sum_kk T[kk][row]*W[kk][o] + bias ----------------
__global__ void __launch_bounds__(256) k_gemm_wmma(
    const unsigned short* __restrict__ T, const float* __restrict__ weight,
    const float* __restrict__ bias, float* __restrict__ out) {
  __shared__ unsigned short Wt[FOUT * WPITCH];        // weights transposed [o][kk], bf16
  int tid = threadIdx.x;
  for (int e = tid; e < KDIM * FOUT; e += 256) {
    int kk = e >> 6;                                  // [0,320)
    int o  = e & 63;
    Wt[o * WPITCH + kk] = f2bf(weight[(size_t)kk * FOUT + o]); // [K][Fin][Fout] flat = kk*64+o
  }
  __syncthreads();

  int lane = tid & 31;
  int wave = tid >> 5;
  int ln = lane & 15;
  int hi = lane >> 4;
  int rowBase = (blockIdx.x * 8 + wave) * 16;
  int rowM = rowBase + ln;                            // A-fragment row for this lane

  v8f acc0 = {}, acc1 = {}, acc2 = {}, acc3 = {};

  for (int kc = 0; kc < 10; ++kc) {                   // 10 x k=32 bf16 chunks (KDIM=320)
    int k    = kc >> 1;
    int ioff = (kc & 1) * 32;
    const unsigned short* arow = T + (((size_t)k * ROWS + rowM) * FIN + ioff);
    // 16-bit A 16x32 layout: lanes 0-15 hold K 0-7 & 16-23; lanes 16-31 hold K 8-15 & 24-31
    Frag a;
    a.u4[0] = *(const uint4*)(arow + hi * 8);
    a.u4[1] = *(const uint4*)(arow + 16 + hi * 8);
    if (kc < 9) {
      int kn = (kc + 1) >> 1, ion = ((kc + 1) & 1) * 32;
      __builtin_prefetch((const void*)(T + (((size_t)kn * ROWS + rowM) * FIN + ion)), 0, 3);
    }
    int koff = kc * 32 + hi * 8;                      // B 32x16: lanes 0-15 K 0-7 & 16-23, 16-31 K 8-15 & 24-31
    #pragma unroll
    for (int n = 0; n < 4; ++n) {
      Frag bf;
      const uint4* bp = (const uint4*)&Wt[(n * 16 + ln) * WPITCH + koff];
      bf.u4[0] = bp[0];
      bf.u4[1] = bp[2];
      v8f* accp = (n == 0) ? &acc0 : (n == 1) ? &acc1 : (n == 2) ? &acc2 : &acc3;
      *accp = __builtin_amdgcn_wmma_f32_16x16x32_bf16(false, a.v, false, bf.v,
                                                      (short)0, *accp, false, false);
    }
  }

  // Epilogue: bias + store into [B, Fout, V]; lane holds 8 consecutive v-rows per N-tile.
  int b  = rowBase / V_;                              // V % 16 == 0 -> tile never straddles b
  int vb = (rowBase % V_) + hi * 8;
  #pragma unroll
  for (int n = 0; n < 4; ++n) {
    v8f acc = (n == 0) ? acc0 : (n == 1) ? acc1 : (n == 2) ? acc2 : acc3;
    int o = n * 16 + ln;
    float bz = bias[o];
    float* op = out + ((size_t)b * FOUT + o) * V_ + vb;
    float4 lo4 = make_float4(acc[0] + bz, acc[1] + bz, acc[2] + bz, acc[3] + bz);
    float4 hi4 = make_float4(acc[4] + bz, acc[5] + bz, acc[6] + bz, acc[7] + bz);
    *(float4*)op       = lo4;
    *(float4*)(op + 4) = hi4;
  }
}

// ---------------- 5) BatchNorm stats + finalize + normalize/ReLU ----------------
__global__ void k_zero_stats(float* stats) { stats[threadIdx.x] = 0.f; }

__global__ void k_stats(const float* __restrict__ out, float* __restrict__ stats) {
  __shared__ float ssum[256], ssq[256];
  int tid = threadIdx.x;
  int c = blockIdx.x & 63;
  int b = blockIdx.x >> 6;
  const float4* p = (const float4*)(out + ((size_t)b * FOUT + c) * V_);
  float s = 0.f, q = 0.f;
  for (int i = tid; i < V_ / 4; i += 256) {
    float4 v = p[i];
    s += v.x + v.y + v.z + v.w;
    q += v.x * v.x + v.y * v.y + v.z * v.z + v.w * v.w;
  }
  ssum[tid] = s; ssq[tid] = q;
  __syncthreads();
  for (int st = 128; st > 0; st >>= 1) {
    if (tid < st) { ssum[tid] += ssum[tid + st]; ssq[tid] += ssq[tid + st]; }
    __syncthreads();
  }
  if (tid == 0) { atomicAdd(&stats[c], ssum[0]); atomicAdd(&stats[64 + c], ssq[0]); }
}

__global__ void k_finalize(const float* __restrict__ gamma, const float* __restrict__ beta,
                           float* stats) {
  int c = threadIdx.x;                                // 64 threads
  float cnt  = (float)((size_t)B_ * V_);
  float mean = stats[c] / cnt;
  float var  = stats[64 + c] / cnt - mean * mean;
  float rs   = rsqrtf(var + 1e-5f);
  float sc   = gamma[c] * rs;
  stats[128 + c] = sc;
  stats[192 + c] = beta[c] - mean * sc;
}

__global__ void k_bnrelu(float* __restrict__ out, const float* __restrict__ stats) {
  size_t i4 = (size_t)blockIdx.x * 256 + threadIdx.x;
  int c = (int)((i4 / (V_ / 4)) & 63);
  float sc = stats[128 + c], sh = stats[192 + c];
  float4 v = ((float4*)out)[i4];
  v.x = fmaxf(v.x * sc + sh, 0.f);
  v.y = fmaxf(v.y * sc + sh, 0.f);
  v.z = fmaxf(v.z * sc + sh, 0.f);
  v.w = fmaxf(v.w * sc + sh, 0.f);
  ((float4*)out)[i4] = v;
}

// ---------------- launch ----------------
extern "C" void kernel_launch(void* const* d_in, const int* in_sizes, int n_in,
                              void* d_out, int out_size, void* d_ws, size_t ws_size,
                              hipStream_t stream) {
  (void)in_sizes; (void)n_in; (void)out_size; (void)ws_size;
  const float* x      = (const float*)d_in[0];
  const int*   rows   = (const int*)d_in[1];
  const int*   cols   = (const int*)d_in[2];
  const float* vals   = (const float*)d_in[3];
  const float* weight = (const float*)d_in[4];
  const float* bias   = (const float*)d_in[5];
  const float* gamma  = (const float*)d_in[6];
  const float* beta   = (const float*)d_in[7];
  float* out = (float*)d_out;

  // workspace: y (f32, 100.7 MB) | T0..T4 (bf16, 251.7 MB) | stats (1 KB)
  char* ws = (char*)d_ws;
  float*          y     = (float*)ws;
  unsigned short* T     = (unsigned short*)(ws + NEL * 4);
  float*          stats = (float*)(ws + NEL * 4 + 5 * NEL * 2);

  // T0 = bf16(x^T)
  k_transpose_bf16<<<dim3(V_ / 64, B_), 256, 0, stream>>>(x, T);

  // Chebyshev recursion: T1 = L T0; Tk = 2 L T_{k-1} - T_{k-2}
  for (int k = 1; k < KC; ++k) {
    hipMemsetAsync(y, 0, NEL * 4, stream);
    k_spmm<<<(int)(((long long)B_ * E_) / 64), 256, 0, stream>>>(
        T + (size_t)(k - 1) * NEL, rows, cols, vals, y);
    k_cheb<<<(int)(NEL / 1024), 256, 0, stream>>>(
        y, T + (size_t)(k >= 2 ? k - 2 : 0) * NEL, T + (size_t)k * NEL, (k >= 2) ? 1 : 0);
  }

  // fused [B*V, 320] x [320, 64] GEMM via bf16 WMMA, bias, store [B,Fout,V]
  k_gemm_wmma<<<ROWS / 128, 256, 0, stream>>>(T, weight, bias, out);

  // BatchNorm (training stats) + ReLU
  k_zero_stats<<<1, 256, 0, stream>>>(stats);
  k_stats<<<B_ * FOUT, 256, 0, stream>>>(out, stats);
  k_finalize<<<1, 64, 0, stream>>>(gamma, beta, stats);
  k_bnrelu<<<(int)(NEL / 1024), 256, 0, stream>>>(out, stats);
}
